// FDKN_2680059593432
// MI455X (gfx1250) — matrix operand order repaired and checked by reading, not compiled, not run on hardware
//
#include <hip/hip_runtime.h>
#include <hip/hip_bf16.h>

// ---------------------------------------------------------------------------
// FDKN on gfx1250: implicit-GEMM convs via v_wmma_f32_16x16x32_f16 (wave32),
// f16 activations / folded weights, f32 accumulation, fused deformable sample.
// ---------------------------------------------------------------------------

typedef __attribute__((ext_vector_type(16))) _Float16    v16h;
typedef __attribute__((ext_vector_type(8)))  float       v8f;
typedef __attribute__((ext_vector_type(4)))  unsigned int v4u;

#define P_LOW   32768          // 2 * 128 * 128 low-res positions
#define HIRES   262144         // 512*512 per batch
#define NPIX    (2 * HIRES)

enum { ACT_RELU = 0, ACT_SIGMOID = 1, ACT_NONE = 2 };

// ---------------------------------------------------------------------------
// Fold BN into conv weights, reorder OIHW -> A[M=cout][K = tap*cin+ci] (f16),
// zero-padded to kpad (multiple of 32). Also emits folded f32 bias.
// ---------------------------------------------------------------------------
__global__ void fold_weights(const float* __restrict__ w, const float* __restrict__ b,
                             const float* __restrict__ g, const float* __restrict__ be,
                             const float* __restrict__ m, const float* __restrict__ v,
                             int cout, int cin, int ks, int kpad,
                             _Float16* __restrict__ wA, float* __restrict__ bias)
{
    int idx = blockIdx.x * blockDim.x + threadIdx.x;
    int total = cout * kpad;
    if (idx >= total) return;
    int o = idx / kpad;
    int k = idx % kpad;
    float s = 1.0f;
    if (g) s = g[o] * rsqrtf(v[o] + 1e-5f);
    float val = 0.0f;
    int ktot = ks * ks * cin;
    if (k < ktot) {
        int tap = k / cin;
        int ci  = k % cin;
        val = w[(o * cin + ci) * (ks * ks) + tap] * s;
    }
    wA[idx] = (_Float16)val;
    if (k == 0) {
        float bb = b[o];
        if (g) bb = (bb - m[o]) * s + be[o];
        bias[o] = bb;
    }
}

// ---------------------------------------------------------------------------
// Grayscale + space-to-channel resample: act[p][c] with c = i*4+j,
// value = src[b, ho*4+i, wo*4+j].
// ---------------------------------------------------------------------------
__global__ __launch_bounds__(256) void prep_kernel(const float* __restrict__ image,
                                                   const float* __restrict__ depth,
                                                   _Float16* __restrict__ act, int gray)
{
    int idx = blockIdx.x * blockDim.x + threadIdx.x;
    if (idx >= P_LOW * 16) return;
    int p = idx >> 4, c = idx & 15;
    int b = p >> 14, rem = p & 16383;
    int ho = rem >> 7, wo = rem & 127;
    int i = c >> 2, j = c & 3;
    int y = ho * 4 + i, x = wo * 4 + j;
    float val;
    if (gray) {
        const float* base = image + (size_t)b * 3 * HIRES + (size_t)y * 512 + x;
        val = 0.299f * base[0] + 0.587f * base[HIRES] + 0.114f * base[2 * HIRES];
    } else {
        val = depth[(size_t)b * HIRES + (size_t)y * 512 + x];
    }
    act[(size_t)p * 16 + c] = (_Float16)val;
}

// ---------------------------------------------------------------------------
// Implicit-GEMM conv (KSxKS, pad KS/2) as D[M=COUT][N=positions] via WMMA.
// Block: 256 threads = 8 waves, tile = NPOS positions x COUT.
// Fast path (CIN % 32 == 0): a 32-wide K chunk never crosses a tap boundary,
// so im2col staging is one b128 global load + one b128 LDS store per slot.
// LDS rows padded to 80B (16B-aligned) to spread fragment reads over banks.
// ---------------------------------------------------------------------------
template <int CIN, int COUT, int KS, int NPOS, int ACT, bool F32OUT, bool MUL>
__global__ __launch_bounds__(256) void conv_wmma(const _Float16* __restrict__ act_in,
                                                 const _Float16* __restrict__ wA,
                                                 const float* __restrict__ bias,
                                                 _Float16* __restrict__ out16,
                                                 float* __restrict__ out32,
                                                 const float* __restrict__ other)
{
    constexpr int KTOT   = KS * KS * CIN;
    constexpr int KT     = (KTOT + 31) / 32;
    constexpr int KPAD   = KT * 32;
    constexpr int MT     = COUT / 16;
    constexpr int NT     = NPOS / 16;
    constexpr int TILES  = MT * NT;
    constexpr int NWAVES = 8;
    constexpr int NACC   = (TILES + NWAVES - 1) / NWAVES;
    constexpr int LDSR   = 40;                    // halves per LDS row (80B stride)
    constexpr bool FAST  = (CIN % 32) == 0;       // chunk == contiguous 32ch of one tap

    __shared__ __align__(16) _Float16 ldsA[COUT * LDSR];   // weights chunk [COUT][32]
    __shared__ __align__(16) _Float16 ldsB[NPOS * LDSR];   // im2col chunk  [NPOS][32]

    const int tid  = threadIdx.x;
    const int wave = tid >> 5;
    const int lane = tid & 31;
    const int posBase = blockIdx.x * NPOS;

    v8f acc[NACC];
#pragma unroll
    for (int i = 0; i < NACC; ++i) {
        v8f z = {0.f, 0.f, 0.f, 0.f, 0.f, 0.f, 0.f, 0.f};
        acc[i] = z;
    }

    for (int kc = 0; kc < KT; ++kc) {
        // ---- stage A chunk: COUT rows x 32 halves (16B vectors) ----
        for (int q = tid; q < COUT * 4; q += 256) {
            int mrow = q >> 2, part = q & 3;
            *(v4u*)(ldsA + mrow * LDSR + part * 8) =
                *(const v4u*)(wA + (size_t)mrow * KPAD + kc * 32 + part * 8);
        }
        if ((tid >> 2) < COUT && kc + 1 < KT)   // nudge next weight chunk into caches
            __builtin_prefetch(wA + (size_t)(tid >> 2) * KPAD + (kc + 1) * 32, 0, 1);

        // ---- stage B chunk: on-the-fly im2col ----
        if (FAST) {
            const int kb  = kc * 32;
            const int tap = kb / CIN;             // scalar; 0 for KS==1
            const int cib = kb % CIN;
            const int ky  = (KS == 3) ? (tap / 3 - 1) : 0;
            const int kx  = (KS == 3) ? (tap % 3 - 1) : 0;
#pragma unroll
            for (int slot = tid; slot < NPOS * 4; slot += 256) {
                int n = slot >> 2, part = slot & 3;
                int pos = posBase + n;
                int b   = pos >> 14;
                int rem = pos & 16383;
                int y   = rem >> 7, x = rem & 127;
                int yy  = y + ky,   xx = x + kx;
                v4u val = {0u, 0u, 0u, 0u};
                if (KS == 1 || (yy >= 0 && yy < 128 && xx >= 0 && xx < 128))
                    val = *(const v4u*)(act_in +
                          (size_t)((b << 14) + (yy << 7) + xx) * CIN + cib + part * 8);
                *(v4u*)(ldsB + n * LDSR + part * 8) = val;
            }
        } else {  // generic per-element path (layer 1 only: CIN=16, K crosses taps)
            for (int q = tid; q < NPOS * 32; q += 256) {
                int n = q >> 5, kk = q & 31;
                int k = kc * 32 + kk;
                _Float16 val = (_Float16)0.0f;
                if (k < KTOT) {
                    int tap = k / CIN;
                    int ci  = k % CIN;
                    int pos = posBase + n;
                    int b   = pos >> 14;
                    int rem = pos & 16383;
                    int y   = rem >> 7, x = rem & 127;
                    int yy  = y + tap / 3 - 1;
                    int xx  = x + tap % 3 - 1;
                    if (yy >= 0 && yy < 128 && xx >= 0 && xx < 128)
                        val = act_in[(size_t)((b << 14) + (yy << 7) + xx) * CIN + ci];
                }
                ldsB[n * LDSR + kk] = val;
            }
        }
        __syncthreads();

        // ---- WMMA over this chunk ----
#pragma unroll
        for (int li = 0; li < NACC; ++li) {
            int t = wave + li * NWAVES;
            if (t >= TILES) break;
            int mt = t / NT, nt = t % NT;
            // A fragment: row = lane%16; K runs {kb0..kb0+7, kb0+16..kb0+23}
            int row = mt * 16 + (lane & 15);
            int kb0 = (lane < 16) ? 0 : 8;
            union { v16h h; v4u u[2]; } ua;
            ua.u[0] = *(const v4u*)(ldsA + row * LDSR + kb0);
            ua.u[1] = *(const v4u*)(ldsA + row * LDSR + kb0 + 16);
            // B fragment: col = lane%16; K runs {kbB..kbB+7, kbB+16..kbB+23}
            int col = nt * 16 + (lane & 15);
            int kbB = (lane < 16) ? 0 : 8;
            union { v16h h; v4u u[2]; } ub;
            ub.u[0] = *(const v4u*)(ldsB + col * LDSR + kbB);
            ub.u[1] = *(const v4u*)(ldsB + col * LDSR + kbB + 16);
            acc[li] = __builtin_amdgcn_wmma_f32_16x16x32_f16(
                false, ua.h, false, ub.h, (short)0, acc[li], false, false);
        }
        __syncthreads();
    }

    // ---- epilogue: bias + activation (+ elementwise product), store [pos][COUT] ----
#pragma unroll
    for (int li = 0; li < NACC; ++li) {
        int t = wave + li * NWAVES;
        if (t >= TILES) break;
        int mt = t / NT, nt = t % NT;
        int col   = nt * 16 + (lane & 15);
        int pos   = posBase + col;
        int mbase = mt * 16 + ((lane < 16) ? 0 : 8);
        union { v4u u[2]; float f[8]; } bs;
        bs.u[0] = *(const v4u*)(bias + mbase);
        bs.u[1] = *(const v4u*)(bias + mbase + 4);
        if (F32OUT) {
            union { v4u u[2]; float f[8]; } ot, st;
            if (MUL) {
                ot.u[0] = *(const v4u*)(other + (size_t)pos * COUT + mbase);
                ot.u[1] = *(const v4u*)(other + (size_t)pos * COUT + mbase + 4);
            }
#pragma unroll
            for (int r = 0; r < 8; ++r) {
                float vv = acc[li][r] + bs.f[r];
                if (ACT == ACT_RELU)         vv = fmaxf(vv, 0.0f);
                else if (ACT == ACT_SIGMOID) vv = 1.0f / (1.0f + __expf(-vv));
                if (MUL) vv *= ot.f[r];
                st.f[r] = vv;
            }
            *(v4u*)(out32 + (size_t)pos * COUT + mbase)     = st.u[0];
            *(v4u*)(out32 + (size_t)pos * COUT + mbase + 4) = st.u[1];
        } else {
            union { v4u u; _Float16 h[8]; } st;
#pragma unroll
            for (int r = 0; r < 8; ++r) {
                float vv = acc[li][r] + bs.f[r];
                if (ACT == ACT_RELU)         vv = fmaxf(vv, 0.0f);
                else if (ACT == ACT_SIGMOID) vv = 1.0f / (1.0f + __expf(-vv));
                st.h[r] = (_Float16)vv;
            }
            *(v4u*)(out16 + (size_t)pos * COUT + mbase) = st.u;
        }
    }
}

// ---------------------------------------------------------------------------
// Deformable sampling fuse: pixel-shuffle gather of weights/offsets, weight
// mean-subtraction, 3x3 bilinear samples on the 15x15 patch canvas (read
// straight from depth with canvas+image bounds), dot, + depth residual.
// Coordinates simplify: patch x = vals[i]+off0-0.5, y = vals[j]+off1-0.5.
// ---------------------------------------------------------------------------
__device__ __forceinline__ float samp_patch(const float* __restrict__ dimg,
                                            int yc, int xc, int yy, int xx)
{
    if (yy < 0 || yy > 14 || xx < 0 || xx > 14) return 0.0f;   // canvas (zero pad)
    int Y = yc - 7 + yy, X = xc - 7 + xx;
    if (Y < 0 || Y >= 512 || X < 0 || X >= 512) return 0.0f;   // unfold zero pad
    return dimg[(size_t)Y * 512 + X];
}

__global__ __launch_bounds__(256) void fuse_kernel(const float* __restrict__ wk,  // [P][144]
                                                   const float* __restrict__ wo,  // [P][288]
                                                   const float* __restrict__ depth,
                                                   float* __restrict__ out)
{
    int idx = blockIdx.x * blockDim.x + threadIdx.x;
    if (idx >= NPIX) return;
    int b = idx >> 18;
    int rem = idx & (HIRES - 1);
    int y = rem >> 9, x = rem & 511;
    int p = (b << 14) + ((y >> 2) << 7) + (x >> 2);
    int s = ((y & 3) << 2) + (x & 3);

    const float* wkp = wk + (size_t)p * 144 + s;
    const float* wop = wo + (size_t)p * 288 + s;

    float w[9];
    float mean = 0.0f;
#pragma unroll
    for (int r = 0; r < 9; ++r) { w[r] = wkp[r * 16]; mean += w[r]; }
    mean *= (1.0f / 9.0f);

    const float* dimg = depth + (size_t)b * HIRES;
    float acc = 0.0f;
#pragma unroll
    for (int i = 0; i < 3; ++i) {
#pragma unroll
        for (int j = 0; j < 3; ++j) {
            int r = i * 3 + j;
            float off0 = wop[(r * 2 + 0) * 16];
            float off1 = wop[(r * 2 + 1) * 16];
            float px = (float)(7 + i) + off0 - 0.5f;   // grid[i,j] = (vals[i], vals[j])
            float py = (float)(7 + j) + off1 - 0.5f;
            float fx = floorf(px), fy = floorf(py);
            float wx = px - fx, wy = py - fy;
            int x0 = (int)fx, y0 = (int)fy;
            float v00 = samp_patch(dimg, y, x, y0,     x0);
            float v01 = samp_patch(dimg, y, x, y0,     x0 + 1);
            float v10 = samp_patch(dimg, y, x, y0 + 1, x0);
            float v11 = samp_patch(dimg, y, x, y0 + 1, x0 + 1);
            float sv = v00 * (1.0f - wx) * (1.0f - wy) + v01 * wx * (1.0f - wy)
                     + v10 * (1.0f - wx) * wy          + v11 * wx * wy;
            acc += sv * (w[r] - mean);
        }
    }
    out[idx] = acc + dimg[(size_t)y * 512 + x];
}

// ---------------------------------------------------------------------------
// Host orchestration
// ---------------------------------------------------------------------------
template <int CIN, int COUT, int KS, int NPOS, int ACT, bool F32OUT, bool MUL>
static void launch_conv(const _Float16* in, const _Float16* wA, const float* bias,
                        _Float16* o16, float* o32, const float* other, hipStream_t s)
{
    conv_wmma<CIN, COUT, KS, NPOS, ACT, F32OUT, MUL>
        <<<P_LOW / NPOS, 256, 0, s>>>(in, wA, bias, o16, o32, other);
}

static void launch_fold(const void* w, const void* b, const void* g, const void* be,
                        const void* m, const void* v, int cout, int cin, int ks, int kpad,
                        _Float16* wA, float* bias, hipStream_t s)
{
    int total = cout * kpad;
    fold_weights<<<(total + 255) / 256, 256, 0, s>>>(
        (const float*)w, (const float*)b, (const float*)g, (const float*)be,
        (const float*)m, (const float*)v, cout, cin, ks, kpad, wA, bias);
}

extern "C" void kernel_launch(void* const* d_in, const int* in_sizes, int n_in,
                              void* d_out, int out_size, void* d_ws, size_t ws_size,
                              hipStream_t stream)
{
    (void)in_sizes; (void)n_in; (void)out_size; (void)ws_size;
    const float* image = (const float*)d_in[0];
    const float* depth = (const float*)d_in[1];

    uintptr_t cur = (uintptr_t)d_ws;
    auto carve = [&](size_t bytes) -> void* {
        cur = (cur + 255) & ~(uintptr_t)255;
        void* p = (void*)cur;
        cur += bytes;
        return p;
    };

    _Float16* act0 = (_Float16*)carve((size_t)P_LOW * 16 * 2);
    _Float16* bufA = (_Float16*)carve((size_t)P_LOW * 128 * 2);
    _Float16* bufB = (_Float16*)carve((size_t)P_LOW * 128 * 2);
    float*    wk   = (float*)carve((size_t)P_LOW * 144 * 4);   // sigmoid(im)*sigmoid(dp)
    float*    wof  = (float*)carve((size_t)P_LOW * 288 * 4);   // imo*dpo

    _Float16* wA1 = (_Float16*)carve(32  * 160  * 2);
    _Float16* wA2 = (_Float16*)carve(32  * 288  * 2);
    _Float16* wA3 = (_Float16*)carve(64  * 288  * 2);
    _Float16* wA4 = (_Float16*)carve(64  * 576  * 2);
    _Float16* wA5 = (_Float16*)carve(128 * 576  * 2);
    _Float16* wA6 = (_Float16*)carve(128 * 1152 * 2);
    _Float16* wAw = (_Float16*)carve(144 * 128  * 2);
    _Float16* wAo = (_Float16*)carve(288 * 128  * 2);
    float* bi1 = (float*)carve(32 * 4);
    float* bi2 = (float*)carve(32 * 4);
    float* bi3 = (float*)carve(64 * 4);
    float* bi4 = (float*)carve(64 * 4);
    float* bi5 = (float*)carve(128 * 4);
    float* bi6 = (float*)carve(128 * 4);
    float* biw = (float*)carve(144 * 4);
    float* bio = (float*)carve(288 * 4);

    for (int tower = 0; tower < 2; ++tower) {
        const int T = 2 + tower * 28;   // im_params at 2..29, dp_params at 30..57
        // fold weights (BN on layers 1,3,5)
        launch_fold(d_in[T+0],  d_in[T+1],  d_in[T+2],  d_in[T+3],  d_in[T+4],  d_in[T+5],
                    32, 16, 3, 160, wA1, bi1, stream);
        launch_fold(d_in[T+6],  d_in[T+7],  nullptr, nullptr, nullptr, nullptr,
                    32, 32, 3, 288, wA2, bi2, stream);
        launch_fold(d_in[T+8],  d_in[T+9],  d_in[T+10], d_in[T+11], d_in[T+12], d_in[T+13],
                    64, 32, 3, 288, wA3, bi3, stream);
        launch_fold(d_in[T+14], d_in[T+15], nullptr, nullptr, nullptr, nullptr,
                    64, 64, 3, 576, wA4, bi4, stream);
        launch_fold(d_in[T+16], d_in[T+17], d_in[T+18], d_in[T+19], d_in[T+20], d_in[T+21],
                    128, 64, 3, 576, wA5, bi5, stream);
        launch_fold(d_in[T+22], d_in[T+23], nullptr, nullptr, nullptr, nullptr,
                    128, 128, 3, 1152, wA6, bi6, stream);
        launch_fold(d_in[T+24], d_in[T+25], nullptr, nullptr, nullptr, nullptr,
                    144, 128, 1, 128, wAw, biw, stream);
        launch_fold(d_in[T+26], d_in[T+27], nullptr, nullptr, nullptr, nullptr,
                    288, 128, 1, 128, wAo, bio, stream);

        prep_kernel<<<(P_LOW * 16 + 255) / 256, 256, 0, stream>>>(image, depth, act0,
                                                                  tower == 0 ? 1 : 0);

        launch_conv<16,  32,  3, 128, ACT_RELU, false, false>(act0, wA1, bi1, bufA, nullptr, nullptr, stream);
        launch_conv<32,  32,  3, 128, ACT_RELU, false, false>(bufA, wA2, bi2, bufB, nullptr, nullptr, stream);
        launch_conv<32,  64,  3, 128, ACT_RELU, false, false>(bufB, wA3, bi3, bufA, nullptr, nullptr, stream);
        launch_conv<64,  64,  3, 128, ACT_RELU, false, false>(bufA, wA4, bi4, bufB, nullptr, nullptr, stream);
        launch_conv<64,  128, 3, 128, ACT_RELU, false, false>(bufB, wA5, bi5, bufA, nullptr, nullptr, stream);
        launch_conv<128, 128, 3, 128, ACT_RELU, false, false>(bufA, wA6, bi6, bufB, nullptr, nullptr, stream);

        if (tower == 0) {
            launch_conv<128, 144, 1, 64, ACT_SIGMOID, true, false>(bufB, wAw, biw, nullptr, wk,  nullptr, stream);
            launch_conv<128, 288, 1, 64, ACT_NONE,    true, false>(bufB, wAo, bio, nullptr, wof, nullptr, stream);
        } else {  // multiply dp-tower heads into the stored im-tower heads
            launch_conv<128, 144, 1, 64, ACT_SIGMOID, true, true >(bufB, wAw, biw, nullptr, wk,  wk,  stream);
            launch_conv<128, 288, 1, 64, ACT_NONE,    true, true >(bufB, wAo, bio, nullptr, wof, wof, stream);
        }
    }

    fuse_kernel<<<(NPIX + 255) / 256, 256, 0, stream>>>(wk, wof, depth, (float*)d_out);
}
